// TriangleAttentionCore_40527311405657
// MI455X (gfx1250) — matrix operand order repaired and checked by reading, not compiled, not run on hardware
//
#include <hip/hip_runtime.h>

// Problem constants (from reference)
#define Nn   256
#define Qq   256
#define CZc  128
#define Hh   4
#define CHc  32
#define Dd   128            // H*CH
#define MTOT (Nn * Qq)      // 65536 rows

typedef __attribute__((ext_vector_type(16))) __bf16    v16bf;
typedef __attribute__((ext_vector_type(8)))  float     v8f;
typedef __attribute__((ext_vector_type(4)))  unsigned  u32x4;
typedef __attribute__((ext_vector_type(4)))  int       i32x4;
typedef __attribute__((ext_vector_type(4)))  float     f32x4;

union BFrag { u32x4 q[2]; unsigned short s[16]; v16bf v; };

// Native bf16 converts -> v_cvt_*bf16_f32 (RNE in HW) instead of 4-op software rounding
__device__ __forceinline__ unsigned short f2bf(float f) {
  return __builtin_bit_cast(unsigned short, (__bf16)f);
}
__device__ __forceinline__ float bf2f(unsigned short s) {
  return (float)__builtin_bit_cast(__bf16, s);
}

// Async global->LDS copy path (gfx1250); guarded so the build never breaks.
#if __has_builtin(__builtin_amdgcn_global_load_async_to_lds_b128)
#define HAVE_ASYNC_LDS 1
// Builtin signature (from clang diagnostic): (v4i AS1*, v4i AS3*, imm offset, imm cpol)
#define TO_GLOBAL(p) ((__attribute__((address_space(1))) i32x4*)(p))
#define TO_LDS(p)    ((__attribute__((address_space(3))) i32x4*)(p))
__device__ __forceinline__ void wait_async0() {
#if __has_builtin(__builtin_amdgcn_s_wait_asynccnt)
  __builtin_amdgcn_s_wait_asynccnt(0);
#else
  asm volatile("s_wait_asynccnt 0" ::: "memory");
#endif
}
#else
#define HAVE_ASYNC_LDS 0
#endif

// ---------------------------------------------------------------------------
// Kernel 1: input projections.  dst[row][col] = epilogue( x[row][:] @ w[:, col] )
// mode 0: *= 1/sqrt(CH)   (q)      mode 1: identity (k, v)
// mode 2: sigmoid(. + bias[col])   (gate)
// ---------------------------------------------------------------------------
__global__ __launch_bounds__(256)
void proj_kernel(const float* __restrict__ x, const float* __restrict__ w,
                 const float* __restrict__ bias, int mode,
                 unsigned short* __restrict__ dst) {
  __shared__ unsigned short Wsh[128 * 136];   // padded rows -> conflict-free B frags
  const int tid = threadIdx.x;
  #pragma unroll
  for (int i = 0; i < 64; ++i) {
    int idx = tid + i * 256;
    Wsh[(idx >> 7) * 136 + (idx & 127)] = f2bf(w[idx]);
  }
  __syncthreads();

  const int wave = tid >> 5, lane = tid & 31;
  const int rowbase = blockIdx.x * 128 + wave * 16;
  const int lrow  = lane & 15;
  const int khalf = (lane >> 4) << 3;         // 0 or 8

  v8f acc[8];
  #pragma unroll
  for (int t = 0; t < 8; ++t) acc[t] = v8f{0.f,0.f,0.f,0.f,0.f,0.f,0.f,0.f};

  #pragma unroll
  for (int kc = 0; kc < 4; ++kc) {
    const int kbase = kc * 32;
    // A fragment (16x32 bf16): row = rowbase+lrow, K = kbase+khalf+{0..7,16..23}
    BFrag a;
    const float* ap = x + (size_t)(rowbase + lrow) * CZc + kbase + khalf;
    f32x4 f0 = *(const f32x4*)(ap + 0);
    f32x4 f1 = *(const f32x4*)(ap + 4);
    f32x4 f2 = *(const f32x4*)(ap + 16);
    f32x4 f3 = *(const f32x4*)(ap + 20);
    #pragma unroll
    for (int e = 0; e < 4; ++e) {
      a.s[e]      = f2bf(f0[e]);
      a.s[e + 4]  = f2bf(f1[e]);
      a.s[e + 8]  = f2bf(f2[e]);
      a.s[e + 12] = f2bf(f3[e]);
    }
    #pragma unroll
    for (int ct = 0; ct < 8; ++ct) {
      // B fragment (32x16 bf16, column of W): lane = N col, elements = K rows (table order)
      BFrag b;
      #pragma unroll
      for (int s = 0; s < 16; ++s)
        b.s[s] = Wsh[(kbase + khalf + (s & 7) + ((s >> 3) << 4)) * 136 + ct * 16 + lrow];
      acc[ct] = __builtin_amdgcn_wmma_f32_16x16x32_bf16(
          false, a.v, false, b.v, (short)0, acc[ct], false, false);
    }
  }

  const float qscale = 0.17677669529663687f;  // 1/sqrt(32)
  #pragma unroll
  for (int ct = 0; ct < 8; ++ct) {
    #pragma unroll
    for (int j = 0; j < 8; ++j) {
      int row = rowbase + j + khalf;          // C/D layout: VGPR j -> M=j (+8 upper half)
      int col = ct * 16 + lrow;
      float v = acc[ct][j];
      if (mode == 0)      v *= qscale;
      else if (mode == 2) v = 1.f / (1.f + __expf(-(v + bias[col])));
      dst[(size_t)row * Dd + col] = f2bf(v);
    }
  }
}

// ---------------------------------------------------------------------------
// Kernel 2: fused attention core per (n, h, row-half):
//   logits = q k^T + bias_mask + bias_tri ; softmax ; o = (p @ v) * g
// V tile is prefetched into LDS with GLOBAL_LOAD_ASYNC_TO_LDS_B128 so the copy
// overlaps the logits + softmax phase; fenced with s_wait_asynccnt + barrier.
// ---------------------------------------------------------------------------
__global__ __launch_bounds__(256)
void attn_kernel(const unsigned short* __restrict__ qws,
                 const unsigned short* __restrict__ kws,
                 const unsigned short* __restrict__ vws,
                 const unsigned short* __restrict__ gws,
                 const float* __restrict__ bias_mask,
                 const float* __restrict__ bias_tri,
                 unsigned short* __restrict__ ows) {
  __shared__ unsigned short kT[32 * 264];       // [channel][key], padded stride
  __shared__ unsigned short pslab[8][16 * 40];  // per-wave P restage (16 x 32, padded)
#if HAVE_ASYNC_LDS
  __shared__ unsigned short vsh[256 * 40];      // [key][channel], padded stride
#endif

  const int n = blockIdx.x, h = blockIdx.y, rhalf = blockIdx.z;
  const int tid = threadIdx.x;

#if HAVE_ASYNC_LDS
  // Kick off async V-tile copy first: 256 rows x 64B, 4 x b128 per thread.
  {
    const unsigned short* gp = vws + ((size_t)(n * 256 + tid)) * Dd + h * CHc;
    unsigned short* lp = vsh + tid * 40;
    #pragma unroll
    for (int c = 0; c < 4; ++c)
      __builtin_amdgcn_global_load_async_to_lds_b128(
          TO_GLOBAL(gp + c * 8), TO_LDS(lp + c * 8), 0, 0);
  }
#endif

  // Stage k transposed: thread t owns key row t (32 contiguous bf16 = 64B load)
  {
    union { u32x4 q[4]; unsigned short s[32]; } kk;
    const u32x4* kp = (const u32x4*)(kws + ((size_t)(n * 256 + tid)) * Dd + h * CHc);
    kk.q[0] = kp[0]; kk.q[1] = kp[1]; kk.q[2] = kp[2]; kk.q[3] = kp[3];
    #pragma unroll
    for (int c = 0; c < 32; ++c) kT[c * 264 + tid] = kk.s[c];
  }
  __syncthreads();

  const int wave = tid >> 5, lane = tid & 31;
  const int rowbase = rhalf * 128 + wave * 16;
  const int lrow  = lane & 15;
  const int khalf = (lane >> 4) << 3;

  // q A-fragment (CH = 32 = exactly one WMMA K-step); reused for all 16 col tiles
  BFrag qa;
  {
    const unsigned short* qp =
        qws + ((size_t)(n * 256 + rowbase + lrow)) * Dd + h * CHc + khalf;
    qa.q[0] = *(const u32x4*)(qp);
    qa.q[1] = *(const u32x4*)(qp + 16);
  }

  // Logits: 16 column tiles of 16 keys each
  v8f acc[16];
  #pragma unroll
  for (int ct = 0; ct < 16; ++ct) {
    BFrag kb;  // B = k^T: lane = key column, elements = channels (table order)
    #pragma unroll
    for (int s = 0; s < 16; ++s)
      kb.s[s] = kT[(khalf + (s & 7) + ((s >> 3) << 4)) * 264 + ct * 16 + lrow];
    v8f z = v8f{0.f,0.f,0.f,0.f,0.f,0.f,0.f,0.f};
    acc[ct] = __builtin_amdgcn_wmma_f32_16x16x32_bf16(
        false, qa.v, false, kb.v, (short)0, z, false, false);
  }

  // Add biases, find per-row max (rows live in VGPR j across 16-lane groups)
  float mj[8];
  #pragma unroll
  for (int j = 0; j < 8; ++j) mj[j] = -3.4e38f;
  #pragma unroll
  for (int ct = 0; ct < 16; ++ct) {
    int col = ct * 16 + lrow;
    float bm = bias_mask[n * 256 + col];
    #pragma unroll
    for (int j = 0; j < 8; ++j) {
      int row = rowbase + j + khalf;
      float l = acc[ct][j] + bm + bias_tri[((size_t)h * 256 + row) * 256 + col];
      acc[ct][j] = l;
      mj[j] = fmaxf(mj[j], l);
    }
  }
  #pragma unroll
  for (int j = 0; j < 8; ++j) {
    #pragma unroll
    for (int off = 1; off < 16; off <<= 1)
      mj[j] = fmaxf(mj[j], __shfl_xor(mj[j], off, 32));
  }
  float sj[8];
  #pragma unroll
  for (int j = 0; j < 8; ++j) sj[j] = 0.f;
  #pragma unroll
  for (int ct = 0; ct < 16; ++ct) {
    #pragma unroll
    for (int j = 0; j < 8; ++j) {
      float e = __expf(acc[ct][j] - mj[j]);
      acc[ct][j] = e;
      sj[j] += e;
    }
  }
  #pragma unroll
  for (int j = 0; j < 8; ++j) {
    #pragma unroll
    for (int off = 1; off < 16; off <<= 1)
      sj[j] += __shfl_xor(sj[j], off, 32);
    sj[j] = 1.f / sj[j];
  }

#if HAVE_ASYNC_LDS
  wait_async0();     // own wave's async copies landed in LDS
  __syncthreads();   // all waves' copies visible
#endif

  // o = P @ V over 8 key chunks of 32; P restaged via per-wave LDS slab
  v8f oacc[2];
  oacc[0] = v8f{0.f,0.f,0.f,0.f,0.f,0.f,0.f,0.f};
  oacc[1] = v8f{0.f,0.f,0.f,0.f,0.f,0.f,0.f,0.f};
  unsigned short* slab = pslab[wave];
  for (int kc = 0; kc < 8; ++kc) {
    __syncthreads();   // WAR on slab (uniform loop across all waves)
    #pragma unroll
    for (int t = 0; t < 2; ++t) {
      int ct = kc * 2 + t;
      #pragma unroll
      for (int j = 0; j < 8; ++j) {
        int r = j + khalf;
        int c = t * 16 + lrow;
        slab[r * 40 + c] = f2bf(acc[ct][j] * sj[j]);
      }
    }
    __syncthreads();   // RAW on slab
    BFrag pa;          // A-layout read of the 16x32 P chunk
    {
      const unsigned short* pp = slab + lrow * 40 + khalf;
      pa.q[0] = *(const u32x4*)(pp);
      pa.q[1] = *(const u32x4*)(pp + 16);
    }
    #pragma unroll
    for (int t = 0; t < 2; ++t) {
      BFrag vb;        // B = V chunk: lane = channel column, elements = key rows (table order)
      #pragma unroll
      for (int s = 0; s < 16; ++s) {
        int kr = kc * 32 + khalf + (s & 7) + ((s >> 3) << 4);
#if HAVE_ASYNC_LDS
        vb.s[s] = vsh[kr * 40 + t * 16 + lrow];
#else
        vb.s[s] = vws[((size_t)(n * 256 + kr)) * Dd + h * CHc + t * 16 + lrow];
#endif
      }
      oacc[t] = __builtin_amdgcn_wmma_f32_16x16x32_bf16(
          false, pa.v, false, vb.v, (short)0, oacc[t], false, false);
    }
  }

  // Sigmoid-gate epilogue + store bf16 o
  #pragma unroll
  for (int t = 0; t < 2; ++t) {
    #pragma unroll
    for (int j = 0; j < 8; ++j) {
      int row = rowbase + j + khalf;
      int ch  = t * 16 + lrow;
      size_t idx = ((size_t)(n * 256 + row)) * Dd + h * CHc + ch;
      ows[idx] = f2bf(oacc[t][j] * bf2f(gws[idx]));
    }
  }
}

// ---------------------------------------------------------------------------
// Kernel 3: out = o @ Wo + bo  (fp32 output)
// ---------------------------------------------------------------------------
__global__ __launch_bounds__(256)
void outproj_kernel(const unsigned short* __restrict__ x,
                    const float* __restrict__ w,
                    const float* __restrict__ bias,
                    float* __restrict__ out) {
  __shared__ unsigned short Wsh[128 * 136];
  const int tid = threadIdx.x;
  #pragma unroll
  for (int i = 0; i < 64; ++i) {
    int idx = tid + i * 256;
    Wsh[(idx >> 7) * 136 + (idx & 127)] = f2bf(w[idx]);
  }
  __syncthreads();

  const int wave = tid >> 5, lane = tid & 31;
  const int rowbase = blockIdx.x * 128 + wave * 16;
  const int lrow  = lane & 15;
  const int khalf = (lane >> 4) << 3;

  v8f acc[8];
  #pragma unroll
  for (int t = 0; t < 8; ++t) acc[t] = v8f{0.f,0.f,0.f,0.f,0.f,0.f,0.f,0.f};

  #pragma unroll
  for (int kc = 0; kc < 4; ++kc) {
    BFrag a;
    const unsigned short* ap = x + (size_t)(rowbase + lrow) * Dd + kc * 32 + khalf;
    a.q[0] = *(const u32x4*)(ap);
    a.q[1] = *(const u32x4*)(ap + 16);
    #pragma unroll
    for (int ct = 0; ct < 8; ++ct) {
      BFrag b;
      #pragma unroll
      for (int s = 0; s < 16; ++s)
        b.s[s] = Wsh[(kc * 32 + khalf + (s & 7) + ((s >> 3) << 4)) * 136 + ct * 16 + lrow];
      acc[ct] = __builtin_amdgcn_wmma_f32_16x16x32_bf16(
          false, a.v, false, b.v, (short)0, acc[ct], false, false);
    }
  }

  #pragma unroll
  for (int ct = 0; ct < 8; ++ct) {
    #pragma unroll
    for (int j = 0; j < 8; ++j) {
      int row = rowbase + j + khalf;
      int col = ct * 16 + lrow;
      out[(size_t)row * CZc + col] = acc[ct][j] + bias[col];
    }
  }
}

// ---------------------------------------------------------------------------
extern "C" void kernel_launch(void* const* d_in, const int* in_sizes, int n_in,
                              void* d_out, int out_size, void* d_ws, size_t ws_size,
                              hipStream_t stream) {
  const float* q_x       = (const float*)d_in[0];
  const float* kv_x      = (const float*)d_in[1];
  const float* bias_mask = (const float*)d_in[2];
  const float* bias_tri  = (const float*)d_in[3];
  const float* Wq        = (const float*)d_in[4];
  const float* Wk        = (const float*)d_in[5];
  const float* Wv        = (const float*)d_in[6];
  const float* Wg        = (const float*)d_in[7];
  const float* bg        = (const float*)d_in[8];
  const float* Wo        = (const float*)d_in[9];
  const float* bo        = (const float*)d_in[10];
  float* out             = (float*)d_out;

  // Workspace: 5 bf16 buffers of 65536 x 128  (5 * 16 MiB = 80 MiB)
  const size_t elems = (size_t)MTOT * Dd;
  unsigned short* qws = (unsigned short*)d_ws;
  unsigned short* kws = qws + elems;
  unsigned short* vws = kws + elems;
  unsigned short* gws = vws + elems;
  unsigned short* ows = gws + elems;

  dim3 blk(256);
  dim3 gproj(MTOT / 128);
  proj_kernel<<<gproj, blk, 0, stream>>>(q_x,  Wq, nullptr, 0, qws);
  proj_kernel<<<gproj, blk, 0, stream>>>(kv_x, Wk, nullptr, 1, kws);
  proj_kernel<<<gproj, blk, 0, stream>>>(kv_x, Wv, nullptr, 1, vws);
  proj_kernel<<<gproj, blk, 0, stream>>>(q_x,  Wg, bg,      2, gws);

  dim3 gattn(Nn, Hh, 2);
  attn_kernel<<<gattn, blk, 0, stream>>>(qws, kws, vws, gws, bias_mask, bias_tri, ows);

  outproj_kernel<<<gproj, blk, 0, stream>>>(ows, Wo, bo, out);
}
